// MolGraphBlock_22746146800299
// MI455X (gfx1250) — matrix operand order repaired and checked by reading, not compiled, not run on hardware
//
#include <hip/hip_runtime.h>
#include <cstdint>
#include <cstddef>

// ---------------- types for WMMA ----------------
typedef __attribute__((ext_vector_type(16))) __bf16 v16bf;
typedef __attribute__((ext_vector_type(8)))  float  v8f;

struct U16x16 { uint4 lo, hi; };   // 32 bytes = 16 bf16 elements

static __device__ inline unsigned short f2bf(float f) {
  // round-to-nearest-even truncation to bf16
  unsigned int u = __float_as_uint(f);
  unsigned int r = 0x7FFFu + ((u >> 16) & 1u);
  return (unsigned short)((u + r) >> 16);
}

static __device__ inline void atomicMaxF(float* addr, float val) {
  unsigned int* ua = (unsigned int*)addr;
  unsigned int cur = __float_as_uint(*addr);
  while (__uint_as_float(cur) < val) {
    unsigned int prev = atomicCAS(ua, cur, __float_as_uint(val));
    if (prev == cur) break;
    cur = prev;
  }
}

static __device__ inline void edge_sd(const int* __restrict__ srcI,
                                      const int* __restrict__ dstI,
                                      int E, int e, int& s, int& d) {
  if (e < E) { s = srcI[e]; d = dstI[e]; }
  else       { s = e - E;   d = s; }
}

// ---------------- bf16 conversion kernels ----------------
__global__ void cvt_rows_bf16(const float* __restrict__ src,
                              unsigned short* __restrict__ dst,
                              int M, int K, int Kp) {
  long long t = (long long)blockIdx.x * blockDim.x + threadIdx.x;
  long long tot = (long long)M * Kp;
  if (t >= tot) return;
  int k = (int)(t % Kp);
  long long m = t / Kp;
  dst[t] = (k < K) ? f2bf(src[m * K + k]) : (unsigned short)0;
}

__global__ void cvt_wT_bf16(const float* __restrict__ W,    // [K x N] row-major
                            unsigned short* __restrict__ Wt, // [N x Kp] bf16
                            int K, int N, int Kp) {
  long long t = (long long)blockIdx.x * blockDim.x + threadIdx.x;
  long long tot = (long long)N * Kp;
  if (t >= tot) return;
  int k = (int)(t % Kp);
  long long n = t / Kp;
  Wt[t] = (k < K) ? f2bf(W[(long long)k * N + n]) : (unsigned short)0;
}

// ---------------- WMMA GEMM: C[MxN] = A[MxK](bf16) * B, B given as Bt[NxK](bf16)
__global__ void __launch_bounds__(256)
wmma_gemm_bf16(const unsigned short* __restrict__ A,
               const unsigned short* __restrict__ Bt,
               float* __restrict__ C,
               int M, int N, int K) {
  const int lane   = threadIdx.x & 31;
  const int laneLo = lane & 15;
  const int laneHi = lane >> 4;
  const int tilesN = N >> 6;                       // 64-wide tiles
  const int tilesM = (M + 31) >> 5;                // 32-tall tiles
  const int wave   = blockIdx.x * (blockDim.x >> 5) + (threadIdx.x >> 5);
  if (wave >= tilesM * tilesN) return;
  const int tm = (wave / tilesN) * 32;
  const int tn = (wave % tilesN) * 64;

  v8f acc[2][4];
#pragma unroll
  for (int a = 0; a < 2; ++a)
#pragma unroll
    for (int b = 0; b < 4; ++b)
      acc[a][b] = (v8f){0.f,0.f,0.f,0.f,0.f,0.f,0.f,0.f};

  const unsigned short* pA[2];
#pragma unroll
  for (int a = 0; a < 2; ++a) {
    int r = tm + a * 16 + laneLo;
    if (r >= M) r = M - 1;                          // clamp; store is guarded
    pA[a] = A + (size_t)r * K + laneHi * 8;
  }
  const unsigned short* pB[4];
#pragma unroll
  for (int b = 0; b < 4; ++b) {
    int col = tn + b * 16 + laneLo;
    pB[b] = Bt + (size_t)col * K + laneHi * 8;
  }

  for (int k0 = 0; k0 < K; k0 += 32) {
    v16bf af[2], bfr[4];
#pragma unroll
    for (int a = 0; a < 2; ++a) {
      U16x16 t;
      t.lo = *(const uint4*)(pA[a] + k0);
      t.hi = *(const uint4*)(pA[a] + k0 + 16);
      af[a] = __builtin_bit_cast(v16bf, t);
    }
#pragma unroll
    for (int b = 0; b < 4; ++b) {
      U16x16 t;
      t.lo = *(const uint4*)(pB[b] + k0);
      t.hi = *(const uint4*)(pB[b] + k0 + 16);
      bfr[b] = __builtin_bit_cast(v16bf, t);
    }
#pragma unroll
    for (int a = 0; a < 2; ++a)
#pragma unroll
      for (int b = 0; b < 4; ++b)
        acc[a][b] = __builtin_amdgcn_wmma_f32_16x16x32_bf16(
            false, af[a], false, bfr[b], (short)0, acc[a][b], false, false);
  }

#pragma unroll
  for (int a = 0; a < 2; ++a) {
    int rbase = tm + a * 16 + laneHi * 8;
#pragma unroll
    for (int b = 0; b < 4; ++b) {
      int col = tn + b * 16 + laneLo;
#pragma unroll
      for (int r = 0; r < 8; ++r) {
        int row = rbase + r;
        if (row < M) C[(size_t)row * N + col] = acc[a][b][r];
      }
    }
  }
}

// ---------------- elementwise / fill ----------------
__global__ void fill_f32(float* __restrict__ p, float v, long long n) {
  long long t = (long long)blockIdx.x * blockDim.x + threadIdx.x;
  if (t < n) p[t] = v;
}

__global__ void init_bias_rows(float* __restrict__ out, const float* __restrict__ bias,
                               long long M, int D) {
  long long t = (long long)blockIdx.x * blockDim.x + threadIdx.x;
  if (t >= M * D) return;
  out[t] = bias[(int)(t % D)];
}

__global__ void elu_inplace(float* __restrict__ p, long long n) {
  long long t = (long long)blockIdx.x * blockDim.x + threadIdx.x;
  if (t >= n) return;
  float v = p[t];
  p[t] = v > 0.f ? v : (expf(v) - 1.f);
}

__global__ void relu_inplace(float* __restrict__ p, long long n) {
  long long t = (long long)blockIdx.x * blockDim.x + threadIdx.x;
  if (t >= n) return;
  float v = p[t];
  p[t] = v > 0.f ? v : 0.f;
}

// ---------------- GATv2 edge kernels ----------------
__global__ void edge_logits_kernel(const float* __restrict__ xl, const float* __restrict__ xr,
                                   const float* __restrict__ att,
                                   const int* __restrict__ srcI, const int* __restrict__ dstI,
                                   int E, int ET, float* __restrict__ logits, int H, int C) {
  long long t = (long long)blockIdx.x * blockDim.x + threadIdx.x;
  if (t >= (long long)ET * H) return;
  int e = (int)(t / H), h = (int)(t % H);
  int s, d; edge_sd(srcI, dstI, E, e, s, d);
  const float* pl = xl + (long long)s * H * C + (long long)h * C;
  const float* pr = xr + (long long)d * H * C + (long long)h * C;
  const float* pa = att + (long long)h * C;
  float acc = 0.f;
  for (int c = 0; c < C; ++c) {
    float v = pl[c] + pr[c];
    v = v > 0.f ? v : 0.2f * v;     // leaky_relu, slope 0.2
    acc += v * pa[c];
  }
  logits[t] = acc;
}

__global__ void seg_max_kernel(const float* __restrict__ logits,
                               const int* __restrict__ srcI, const int* __restrict__ dstI,
                               int E, int ET, float* __restrict__ mx, int H) {
  long long t = (long long)blockIdx.x * blockDim.x + threadIdx.x;
  if (t >= (long long)ET * H) return;
  int e = (int)(t / H), h = (int)(t % H);
  int s, d; edge_sd(srcI, dstI, E, e, s, d);
  atomicMaxF(&mx[(long long)d * H + h], logits[t]);
}

__global__ void edge_exp_kernel(float* __restrict__ logits,        // in: logits, out: exp
                                const float* __restrict__ mx,
                                const int* __restrict__ srcI, const int* __restrict__ dstI,
                                int E, int ET, float* __restrict__ den, int H) {
  long long t = (long long)blockIdx.x * blockDim.x + threadIdx.x;
  if (t >= (long long)ET * H) return;
  int e = (int)(t / H), h = (int)(t % H);
  int s, d; edge_sd(srcI, dstI, E, e, s, d);
  float ex = expf(logits[t] - mx[(long long)d * H + h]);
  logits[t] = ex;
  atomicAdd(&den[(long long)d * H + h], ex);
}

__global__ void scatter_attn_kernel(const float* __restrict__ xl, const float* __restrict__ ex,
                                    const float* __restrict__ den,
                                    const int* __restrict__ srcI, const int* __restrict__ dstI,
                                    int E, int ET, float* __restrict__ out, int H, int C, int D) {
  long long t = (long long)blockIdx.x * blockDim.x + threadIdx.x;
  if (t >= (long long)ET * D) return;
  int e = (int)(t / D), j = (int)(t % D);
  int s, d; edge_sd(srcI, dstI, E, e, s, d);
  int h = j / C;
  float alpha = ex[(long long)e * H + h] / den[(long long)d * H + h];
  atomicAdd(&out[(long long)d * D + j], alpha * xl[(long long)s * D + j]);
}

// ---------------- GCN kernels ----------------
__global__ void deg_kernel(const int* __restrict__ srcI, const int* __restrict__ dstI,
                           int E, int ET, float* __restrict__ deg) {
  int e = blockIdx.x * blockDim.x + threadIdx.x;
  if (e >= ET) return;
  int s, d; edge_sd(srcI, dstI, E, e, s, d);
  atomicAdd(&deg[d], 1.0f);
}

__global__ void gcn_scatter_kernel(const float* __restrict__ xw, const float* __restrict__ deg,
                                   const int* __restrict__ srcI, const int* __restrict__ dstI,
                                   int E, int ET, float* __restrict__ out, int D) {
  long long t = (long long)blockIdx.x * blockDim.x + threadIdx.x;
  if (t >= (long long)ET * D) return;
  int e = (int)(t / D), j = (int)(t % D);
  int s, d; edge_sd(srcI, dstI, E, e, s, d);
  float norm = rsqrtf(fmaxf(deg[s], 1.f)) * rsqrtf(fmaxf(deg[d], 1.f));
  atomicAdd(&out[(long long)d * D + j], norm * xw[(long long)s * D + j]);
}

// ---------------- pooling ----------------
__global__ void pool_count_kernel(const int* __restrict__ batch, int M, float* __restrict__ cnt) {
  int n = blockIdx.x * blockDim.x + threadIdx.x;
  if (n >= M) return;
  atomicAdd(&cnt[batch[n]], 1.0f);
}

__global__ void pool_elem_kernel(const float* __restrict__ h, const int* __restrict__ batch,
                                 int M, int D, float* __restrict__ gmax, float* __restrict__ gsum) {
  long long t = (long long)blockIdx.x * blockDim.x + threadIdx.x;
  if (t >= (long long)M * D) return;
  int n = (int)(t / D), j = (int)(t % D);
  int g = batch[n];
  float v = h[t];
  atomicMaxF(&gmax[(long long)g * D + j], v);
  atomicAdd(&gsum[(long long)g * D + j], v);
}

__global__ void pool_final_kernel(const float* __restrict__ gmax, const float* __restrict__ gsum,
                                  const float* __restrict__ cnt, int G, int D,
                                  float* __restrict__ out) {
  long long t = (long long)blockIdx.x * blockDim.x + threadIdx.x;
  if (t >= (long long)G * D) return;
  int g = (int)(t / D), j = (int)(t % D);
  out[(long long)g * 2 * D + j]     = gmax[t];
  out[(long long)g * 2 * D + D + j] = gsum[t] / fmaxf(cnt[g], 1.f);
}

// ---------------- host orchestration ----------------
static inline unsigned nblk(long long n) { return (unsigned)((n + 255) / 256); }

extern "C" void kernel_launch(void* const* d_in, const int* in_sizes, int n_in,
                              void* d_out, int out_size, void* d_ws, size_t ws_size,
                              hipStream_t stream) {
  const int M = 50000, E = 150000, ET = 200000, IN = 78;
  const int HID = 640, H1 = 10, C1 = 64, KP1 = 96, G = 1024;

  const float* x   = (const float*)d_in[0];
  const int*   edg = (const int*)  d_in[1];
  const int*   bat = (const int*)  d_in[2];
  const float* Wl1 = (const float*)d_in[3];
  const float* Wr1 = (const float*)d_in[4];
  const float* a1  = (const float*)d_in[5];
  const float* b1  = (const float*)d_in[6];
  const float* Wl2 = (const float*)d_in[7];
  const float* Wr2 = (const float*)d_in[8];
  const float* a2  = (const float*)d_in[9];
  const float* b2  = (const float*)d_in[10];
  const float* Wg  = (const float*)d_in[11];
  const float* bg  = (const float*)d_in[12];
  float* out = (float*)d_out;

  const int* srcI = edg;
  const int* dstI = edg + E;

  // ---- workspace carve-up (256B aligned) ----
  char* ws = (char*)d_ws;
  size_t off = 0;
  auto alloc = [&](size_t bytes) -> void* {
    size_t cur = (off + 255) & ~(size_t)255;
    off = cur + bytes;
    return (void*)(ws + cur);
  };
  unsigned short* xbf = (unsigned short*)alloc((size_t)M * HID * 2);  // bf16 activations (KP1 or HID wide)
  unsigned short* wl  = (unsigned short*)alloc((size_t)HID * HID * 2);
  unsigned short* wr  = (unsigned short*)alloc((size_t)HID * HID * 2);
  float* xl   = (float*)alloc((size_t)M * HID * 4);
  float* xr   = (float*)alloc((size_t)M * HID * 4);
  float* hbuf = (float*)alloc((size_t)M * HID * 4);
  float* lg   = (float*)alloc((size_t)ET * H1 * 4);
  float* mx   = (float*)alloc((size_t)M * H1 * 4);
  float* den  = (float*)alloc((size_t)M * H1 * 4);
  float* deg  = (float*)alloc((size_t)M * 4);
  float* gmax = (float*)alloc((size_t)G * HID * 4);
  float* gsum = (float*)alloc((size_t)G * HID * 4);
  float* gcnt = (float*)alloc((size_t)G * 4);

  const unsigned gemmBlocks = (unsigned)((((M + 31) / 32) * (HID / 64) + 7) / 8);
  const float NEG_INF = -__builtin_huge_valf();

  // ================= Layer 1: GATv2 (10 heads x 64) =================
  cvt_rows_bf16<<<nblk((long long)M * KP1), 256, 0, stream>>>(x, xbf, M, IN, KP1);
  cvt_wT_bf16  <<<nblk((long long)HID * KP1), 256, 0, stream>>>(Wl1, wl, IN, HID, KP1);
  cvt_wT_bf16  <<<nblk((long long)HID * KP1), 256, 0, stream>>>(Wr1, wr, IN, HID, KP1);
  wmma_gemm_bf16<<<gemmBlocks, 256, 0, stream>>>(xbf, wl, xl, M, HID, KP1);
  wmma_gemm_bf16<<<gemmBlocks, 256, 0, stream>>>(xbf, wr, xr, M, HID, KP1);

  edge_logits_kernel<<<nblk((long long)ET * H1), 256, 0, stream>>>(xl, xr, a1, srcI, dstI, E, ET, lg, H1, C1);
  fill_f32<<<nblk((long long)M * H1), 256, 0, stream>>>(mx, NEG_INF, (long long)M * H1);
  fill_f32<<<nblk((long long)M * H1), 256, 0, stream>>>(den, 0.f, (long long)M * H1);
  seg_max_kernel <<<nblk((long long)ET * H1), 256, 0, stream>>>(lg, srcI, dstI, E, ET, mx, H1);
  edge_exp_kernel<<<nblk((long long)ET * H1), 256, 0, stream>>>(lg, mx, srcI, dstI, E, ET, den, H1);
  init_bias_rows<<<nblk((long long)M * HID), 256, 0, stream>>>(hbuf, b1, M, HID);
  scatter_attn_kernel<<<nblk((long long)ET * HID), 256, 0, stream>>>(xl, lg, den, srcI, dstI, E, ET, hbuf, H1, C1, HID);
  elu_inplace<<<nblk((long long)M * HID), 256, 0, stream>>>(hbuf, (long long)M * HID);

  // ================= Layer 2: GATv2 (1 head x 640) =================
  cvt_rows_bf16<<<nblk((long long)M * HID), 256, 0, stream>>>(hbuf, xbf, M, HID, HID);
  cvt_wT_bf16  <<<nblk((long long)HID * HID), 256, 0, stream>>>(Wl2, wl, HID, HID, HID);
  cvt_wT_bf16  <<<nblk((long long)HID * HID), 256, 0, stream>>>(Wr2, wr, HID, HID, HID);
  wmma_gemm_bf16<<<gemmBlocks, 256, 0, stream>>>(xbf, wl, xl, M, HID, HID);
  wmma_gemm_bf16<<<gemmBlocks, 256, 0, stream>>>(xbf, wr, xr, M, HID, HID);

  edge_logits_kernel<<<nblk((long long)ET), 256, 0, stream>>>(xl, xr, a2, srcI, dstI, E, ET, lg, 1, HID);
  fill_f32<<<nblk((long long)M), 256, 0, stream>>>(mx, NEG_INF, (long long)M);
  fill_f32<<<nblk((long long)M), 256, 0, stream>>>(den, 0.f, (long long)M);
  seg_max_kernel <<<nblk((long long)ET), 256, 0, stream>>>(lg, srcI, dstI, E, ET, mx, 1);
  edge_exp_kernel<<<nblk((long long)ET), 256, 0, stream>>>(lg, mx, srcI, dstI, E, ET, den, 1);
  init_bias_rows<<<nblk((long long)M * HID), 256, 0, stream>>>(hbuf, b2, M, HID);
  scatter_attn_kernel<<<nblk((long long)ET * HID), 256, 0, stream>>>(xl, lg, den, srcI, dstI, E, ET, hbuf, 1, HID, HID);

  // ================= GCN =================
  cvt_rows_bf16<<<nblk((long long)M * HID), 256, 0, stream>>>(hbuf, xbf, M, HID, HID);
  cvt_wT_bf16  <<<nblk((long long)HID * HID), 256, 0, stream>>>(Wg, wl, HID, HID, HID);
  wmma_gemm_bf16<<<gemmBlocks, 256, 0, stream>>>(xbf, wl, xl, M, HID, HID);

  fill_f32<<<nblk((long long)M), 256, 0, stream>>>(deg, 0.f, (long long)M);
  deg_kernel<<<nblk((long long)ET), 256, 0, stream>>>(srcI, dstI, E, ET, deg);
  init_bias_rows<<<nblk((long long)M * HID), 256, 0, stream>>>(xr, bg, M, HID);
  gcn_scatter_kernel<<<nblk((long long)ET * HID), 256, 0, stream>>>(xl, deg, srcI, dstI, E, ET, xr, HID);
  relu_inplace<<<nblk((long long)M * HID), 256, 0, stream>>>(xr, (long long)M * HID);

  // ================= Pooling: concat(global_max, global_mean) =================
  fill_f32<<<nblk((long long)G * HID), 256, 0, stream>>>(gmax, NEG_INF, (long long)G * HID);
  fill_f32<<<nblk((long long)G * HID), 256, 0, stream>>>(gsum, 0.f, (long long)G * HID);
  fill_f32<<<nblk((long long)G), 256, 0, stream>>>(gcnt, 0.f, (long long)G);
  pool_count_kernel<<<nblk((long long)M), 256, 0, stream>>>(bat, M, gcnt);
  pool_elem_kernel <<<nblk((long long)M * HID), 256, 0, stream>>>(xr, bat, M, HID, gmax, gsum);
  pool_final_kernel<<<nblk((long long)G * HID), 256, 0, stream>>>(gmax, gsum, gcnt, G, HID, out);
}
